// SpatialAttentionModel_63900523429982
// MI455X (gfx1250) — matrix operand order repaired and checked by reading, not compiled, not run on hardware
//
#include <hip/hip_runtime.h>
#include <hip/hip_bf16.h>

// ---------------------------------------------------------------------------
// CDNA5 / gfx1250 implementation. wave32. WMMA bf16 (f32 accumulate).
// ---------------------------------------------------------------------------

typedef __bf16 bf16_t;
typedef __attribute__((ext_vector_type(16))) __bf16 v16bf;
typedef __attribute__((ext_vector_type(8)))  __bf16 v8bf;
typedef __attribute__((ext_vector_type(8)))  float  v8f;

#define B_      4
#define T_      12
#define N_      3000
#define DH_     32
#define KH_     8
#define KB_     (KH_ * B_)       // 32
#define MEM_    30
#define TOPK_   50
#define MROWS_  (B_ * T_ * N_)   // 144000

__device__ __forceinline__ v8f wmma_bf16(v16bf a, v16bf b, v8f c) {
  return __builtin_amdgcn_wmma_f32_16x16x32_bf16(false, a, false, b, (short)0, c,
                                                 false, false);
}

// A-fragment for V_WMMA_F32_16X16X32_BF16 from row-major bf16 [*, row_stride].
// Lane<16: K = kbase+0..15 ; lane>=16: K = kbase+16..31.
__device__ __forceinline__ v16bf load_a_frag(const bf16_t* A, int row_stride,
                                             int m_base, int k_base, int lane) {
  const int r    = m_base + (lane & 15);
  const int kofs = (lane >> 4) << 4;  // 0 or 16
  const bf16_t* p = A + (size_t)r * row_stride + k_base + kofs;
  v8bf lo = *(const v8bf*)(p);
  v8bf hi = *(const v8bf*)(p + 8);
  v16bf a;
#pragma unroll
  for (int i = 0; i < 8; ++i) { a[i] = lo[i]; a[8 + i] = hi[i]; }
  return a;
}

// Packed B fragment: Bp[((kt*ntiles + nt)*32 + lane)*16 + i]
__device__ __forceinline__ v16bf load_b_packed(const bf16_t* Bp, int ntiles,
                                               int kt, int nt, int lane) {
  return *(const v16bf*)(Bp + (((size_t)(kt * ntiles + nt) * 32 + lane) << 4));
}

// ---------------------------------------------------------------------------
// Weight packing: B fragment layout element(k,n): k = kt*32 + ((lane>>4)<<4)+i,
// n = nt*16 + (lane&15).
// ---------------------------------------------------------------------------
__global__ void pack_qkv_weights(const float* __restrict__ W7,
                                 const float* __restrict__ W8,
                                 const float* __restrict__ W9,
                                 bf16_t* __restrict__ Wp) {
  int tid = blockIdx.x * 256 + threadIdx.x;       // 512*768 elements
  if (tid >= 512 * 768) return;
  int i    = tid & 15;
  int lane = (tid >> 4) & 31;
  int rest = tid >> 9;
  int nt   = rest % 48;
  int kt   = rest / 48;
  int k = kt * 32 + ((lane >> 4) << 4) + i;
  int c = nt * 16 + (lane & 15);                  // 0..767
  const float* W = (c < 256) ? W7 : ((c < 512) ? W8 : W9);
  int cc = c & 255;
  Wp[tid] = (bf16_t)W[k * 256 + cc];
}

__global__ void pack_weight256(const float* __restrict__ W, bf16_t* __restrict__ Wp) {
  int tid = blockIdx.x * 256 + threadIdx.x;       // 256*256 elements
  if (tid >= 256 * 256) return;
  int i    = tid & 15;
  int lane = (tid >> 4) & 31;
  int rest = tid >> 9;
  int nt   = rest % 16;
  int kt   = rest / 16;
  int k = kt * 32 + ((lane >> 4) << 4) + i;
  int n = nt * 16 + (lane & 15);
  Wp[tid] = (bf16_t)W[k * 256 + n];
}

// Xc = [X | STE] as bf16, row-major [MROWS_, 512]
__global__ void build_xc(const float* __restrict__ X, const float* __restrict__ STE,
                         bf16_t* __restrict__ Xc) {
  size_t i = (size_t)blockIdx.x * 256 + threadIdx.x;
  if (i >= (size_t)MROWS_ * 512) return;
  size_t m = i >> 9;
  int c = (int)(i & 511);
  float v = (c < 256) ? X[m * 256 + c] : STE[m * 256 + (c - 256)];
  Xc[i] = (bf16_t)v;
}

// ---------------------------------------------------------------------------
// QKV GEMM: [144000,512]x[512,768] bf16 WMMA. A tile (32x512, 32KB) staged in
// LDS once per block; each wave register-blocks a 32x32 output tile
// (2 M-tiles x 2 N-tiles -> 4 WMMAs per k-step). Fused bias+ReLU epilogue
// writes q/k/v in split-head layout [KB_, T_, N_, 32] bf16.
// Grid (4500, 3), block 256 (8 waves).
// ---------------------------------------------------------------------------
__global__ void __launch_bounds__(256) qkv_gemm_kernel(
    const bf16_t* __restrict__ Xc, const bf16_t* __restrict__ Wp,
    const float* __restrict__ b7, const float* __restrict__ b8,
    const float* __restrict__ b9,
    bf16_t* __restrict__ qb, bf16_t* __restrict__ kbuf, bf16_t* __restrict__ vb) {
  __shared__ __attribute__((aligned(32))) bf16_t As[32 * 512];
  const int tid  = threadIdx.x;
  const int lane = tid & 31;
  const int wave = tid >> 5;
  const int m0   = blockIdx.x * 32;
  // stage A tile: 32x512 bf16 contiguous (row stride == 512) -> 16K elements
  {
    const bf16_t* Ag = Xc + (size_t)m0 * 512;
#pragma unroll
    for (int i = 0; i < 8; ++i) {
      int e = (i * 256 + tid) * 8;
      *(v8bf*)&As[e] = *(const v8bf*)&Ag[e];
    }
  }
  __syncthreads();
  const int n0  = blockIdx.y * 256 + wave * 32;
  const int nt0 = n0 >> 4;
  v8f acc00 = {}, acc01 = {}, acc10 = {}, acc11 = {};
#pragma unroll 2
  for (int kt = 0; kt < 16; ++kt) {
    v16bf a0 = load_a_frag(As, 512, 0, kt * 32, lane);
    v16bf a1 = load_a_frag(As, 512, 16, kt * 32, lane);
    v16bf b0 = load_b_packed(Wp, 48, kt, nt0, lane);
    v16bf b1 = load_b_packed(Wp, 48, kt, nt0 + 1, lane);
    acc00 = wmma_bf16(a0, b0, acc00);
    acc01 = wmma_bf16(a0, b1, acc01);
    acc10 = wmma_bf16(a1, b0, acc10);
    acc11 = wmma_bf16(a1, b1, acc11);
  }
#pragma unroll
  for (int mt = 0; mt < 2; ++mt) {
#pragma unroll
    for (int t2 = 0; t2 < 2; ++t2) {
      v8f acc = mt ? (t2 ? acc11 : acc10) : (t2 ? acc01 : acc00);
      int c      = n0 + t2 * 16 + (lane & 15);   // 0..767
      int sel    = c >> 8;
      int within = c & 255;
      int head   = within >> 5;
      int d      = within & 31;
      float bias = (sel == 0) ? b7[within] : ((sel == 1) ? b8[within] : b9[within]);
      bf16_t* dst = (sel == 0) ? qb : ((sel == 1) ? kbuf : vb);
#pragma unroll
      for (int r = 0; r < 8; ++r) {
        int m   = m0 + mt * 16 + r + ((lane >> 4) << 3);
        int bb  = m / (T_ * N_);
        int rem = m % (T_ * N_);
        int t   = rem / N_;
        int n   = rem % N_;
        int kbH = head * B_ + bb;
        float v = acc[r] + bias;
        v = v > 0.f ? v : 0.f;
        dst[((size_t)((kbH * T_ + t) * N_ + n)) * 32 + d] = (bf16_t)v;
      }
    }
  }
}

// ---------------------------------------------------------------------------
// 256x256 WMMA GEMM with bias (+optional ReLU). A tile (32x256, 16KB) staged
// in LDS; 8 waves each own 32 cols; 2 M-tiles x 2 N-tiles per wave.
// Grid 4500, block 256.
// ---------------------------------------------------------------------------
__global__ void __launch_bounds__(256) gemm_bias_kernel(
    const bf16_t* __restrict__ A, const bf16_t* __restrict__ Bp,
    const float* __restrict__ bias, bf16_t* __restrict__ out_bf,
    float* __restrict__ out_f32, int relu) {
  __shared__ __attribute__((aligned(32))) bf16_t As[32 * 256];
  const int tid  = threadIdx.x;
  const int lane = tid & 31;
  const int wave = tid >> 5;
  const int m0   = blockIdx.x * 32;
  {
    const bf16_t* Ag = A + (size_t)m0 * 256;
#pragma unroll
    for (int i = 0; i < 4; ++i) {
      int e = (i * 256 + tid) * 8;
      *(v8bf*)&As[e] = *(const v8bf*)&Ag[e];
    }
  }
  __syncthreads();
  const int nt0 = wave * 2;
  v8f acc00 = {}, acc01 = {}, acc10 = {}, acc11 = {};
#pragma unroll 2
  for (int kt = 0; kt < 8; ++kt) {
    v16bf a0 = load_a_frag(As, 256, 0, kt * 32, lane);
    v16bf a1 = load_a_frag(As, 256, 16, kt * 32, lane);
    v16bf b0 = load_b_packed(Bp, 16, kt, nt0, lane);
    v16bf b1 = load_b_packed(Bp, 16, kt, nt0 + 1, lane);
    acc00 = wmma_bf16(a0, b0, acc00);
    acc01 = wmma_bf16(a0, b1, acc01);
    acc10 = wmma_bf16(a1, b0, acc10);
    acc11 = wmma_bf16(a1, b1, acc11);
  }
#pragma unroll
  for (int mt = 0; mt < 2; ++mt) {
#pragma unroll
    for (int t2 = 0; t2 < 2; ++t2) {
      v8f acc = mt ? (t2 ? acc11 : acc10) : (t2 ? acc01 : acc00);
      int c = (nt0 + t2) * 16 + (lane & 15);
      float bv = bias[c];
#pragma unroll
      for (int r = 0; r < 8; ++r) {
        int m = m0 + mt * 16 + r + ((lane >> 4) << 3);
        float v = acc[r] + bv;
        if (relu) v = v > 0.f ? v : 0.f;
        if (out_bf) out_bf[(size_t)m * 256 + c] = (bf16_t)v;
        else        out_f32[(size_t)m * 256 + c] = v;
      }
    }
  }
}

// ---------------------------------------------------------------------------
// Memory-node scores + top-50 (softmax skipped: monotonic, only indices used).
// One block per (kbH*T + t, mem). Scores in LDS, 50 argmax-reduce rounds.
// ---------------------------------------------------------------------------
__global__ void __launch_bounds__(256) scores_topk_kernel(
    const bf16_t* __restrict__ qb, const bf16_t* __restrict__ kbuf,
    const float* __restrict__ node_emb, int* __restrict__ idx_out) {
  __shared__ float ne[64];
  __shared__ float sc[N_];
  __shared__ float rv[256];
  __shared__ int   ri[256];
  const int blk   = blockIdx.x;        // (kbH*T_ + t)*MEM_ + m
  const int m_mem = blk % MEM_;
  const int kt_   = blk / MEM_;
  const int tid   = threadIdx.x;
  if (tid < 64) ne[tid] = node_emb[m_mem * 64 + tid];
  __syncthreads();
  const size_t base = (size_t)kt_ * N_ * 32;
  for (int n = tid; n < N_; n += 256) {
    const v8bf* qp = (const v8bf*)(qb   + base + (size_t)n * 32);
    const v8bf* kp = (const v8bf*)(kbuf + base + (size_t)n * 32);
    float acc = 0.f;
#pragma unroll
    for (int c = 0; c < 4; ++c) {
      v8bf qv = qp[c], kv = kp[c];
#pragma unroll
      for (int i = 0; i < 8; ++i)
        acc += ne[c * 8 + i] * (float)qv[i] + ne[32 + c * 8 + i] * (float)kv[i];
    }
    sc[n] = acc;
  }
  __syncthreads();
  for (int it = 0; it < TOPK_; ++it) {
    float bv = -3.0e38f;
    int bi = 0;
    for (int n = tid; n < N_; n += 256) {
      float v = sc[n];
      if (v > bv) { bv = v; bi = n; }
    }
    rv[tid] = bv; ri[tid] = bi;
    __syncthreads();
    for (int s = 128; s > 0; s >>= 1) {
      if (tid < s && rv[tid + s] > rv[tid]) { rv[tid] = rv[tid + s]; ri[tid] = ri[tid + s]; }
      __syncthreads();
    }
    if (tid == 0) { idx_out[blk * TOPK_ + it] = ri[0]; sc[ri[0]] = -3.0e38f; }
    __syncthreads();
  }
}

// agg = 0, cnt = EPS
__global__ void init_accum(float* __restrict__ agg, float* __restrict__ cnt) {
  size_t i = (size_t)blockIdx.x * 256 + threadIdx.x;
  if (i < (size_t)KB_ * T_ * N_ * 32) agg[i] = 0.f;
  if (i < (size_t)KB_ * T_ * N_)      cnt[i] = 1e-14f;
}

// ---------------------------------------------------------------------------
// Gathered subgraph attention + atomic scatter-add. One block (4 waves) per
// (kbH*T + t, mem). Q,K staged row-major in LDS; V staged TRANSPOSED so the
// WMMA B-fragments are contiguous 32B LDS loads. 50 padded to 64.
// ---------------------------------------------------------------------------
__global__ void __launch_bounds__(128) attn_scatter_kernel(
    const bf16_t* __restrict__ qb, const bf16_t* __restrict__ kbuf,
    const bf16_t* __restrict__ vb, const int* __restrict__ idx,
    float* __restrict__ agg, float* __restrict__ cnt) {
  __shared__ __attribute__((aligned(32))) bf16_t Qs[64 * 32];
  __shared__ __attribute__((aligned(32))) bf16_t Ks[64 * 32];
  __shared__ __attribute__((aligned(32))) bf16_t Vt[32 * 64];
  __shared__ __attribute__((aligned(32))) float  Sm[64 * 64];
  __shared__ __attribute__((aligned(32))) bf16_t Pm[64 * 64];
  __shared__ int sidx[TOPK_];
  const int blk  = blockIdx.x;
  const int kt_  = blk / MEM_;
  const int tid  = threadIdx.x;
  const int lane = tid & 31;
  const int wave = tid >> 5;
  const size_t base = (size_t)kt_ * N_ * 32;
  if (tid < TOPK_) sidx[tid] = idx[blk * TOPK_ + tid];
  __syncthreads();
  // gather (pad rows >=50 with zero)
  for (int e = tid; e < 64 * 32; e += 128) {
    int r = e >> 5, d = e & 31;
    bf16_t qv = (bf16_t)0.f, kv = (bf16_t)0.f, vv = (bf16_t)0.f;
    if (r < TOPK_) {
      size_t off = base + (size_t)sidx[r] * 32 + d;
      qv = qb[off]; kv = kbuf[off]; vv = vb[off];
    }
    Qs[e] = qv; Ks[e] = kv; Vt[d * 64 + r] = vv;
  }
  if (tid < TOPK_)
    unsafeAtomicAdd(&cnt[(size_t)kt_ * N_ + sidx[tid]], 1.0f);
  __syncthreads();
  // S = Q K^T  (16 tiles of 16x16 over 4 waves; Kdim = 32, single WMMA)
  for (int tt = wave * 4; tt < wave * 4 + 4; ++tt) {
    int mt = tt >> 2, jt = tt & 3;
    v16bf a = load_a_frag(Qs, 32, mt * 16, 0, lane);
    v16bf b = *(const v16bf*)&Ks[(jt * 16 + (lane & 15)) * 32 + ((lane >> 4) << 4)];
    v8f c = {};
    c = wmma_bf16(a, b, c);
#pragma unroll
    for (int r = 0; r < 8; ++r)
      Sm[(mt * 16 + r + ((lane >> 4) << 3)) * 64 + jt * 16 + (lane & 15)] = c[r];
  }
  __syncthreads();
  // softmax over j<50 per row, write P (bf16) with zero padding
  if (tid < 64) {
    int r = tid;
    if (r < TOPK_) {
      const float scale = 0.17677669529663687f;  // 1/sqrt(32)
      float mx = -3.0e38f;
      for (int j = 0; j < TOPK_; ++j) {
        float v = Sm[r * 64 + j] * scale;
        if (v > mx) mx = v;
      }
      float sum = 0.f;
      for (int j = 0; j < TOPK_; ++j) {
        float e = __expf(Sm[r * 64 + j] * scale - mx);
        Sm[r * 64 + j] = e;
        sum += e;
      }
      float inv = 1.f / sum;
      for (int j = 0; j < 64; ++j)
        Pm[r * 64 + j] = (bf16_t)((j < TOPK_) ? Sm[r * 64 + j] * inv : 0.f);
    } else {
      for (int j = 0; j < 64; ++j) Pm[r * 64 + j] = (bf16_t)0.f;
    }
  }
  __syncthreads();
  // node_new = P @ V (8 tiles over 4 waves; Kdim = 64 -> 2 WMMA steps), scatter
  for (int tt = wave * 2; tt < wave * 2 + 2; ++tt) {
    int mt = tt >> 1, dt = tt & 1;
    v8f c = {};
#pragma unroll
    for (int kc = 0; kc < 2; ++kc) {
      v16bf a = load_a_frag(Pm, 64, mt * 16, kc * 32, lane);
      v16bf b = *(const v16bf*)&Vt[(dt * 16 + (lane & 15)) * 64 + kc * 32 +
                                   ((lane >> 4) << 4)];
      c = wmma_bf16(a, b, c);
    }
    int d = dt * 16 + (lane & 15);
#pragma unroll
    for (int r = 0; r < 8; ++r) {
      int ksel = mt * 16 + r + ((lane >> 4) << 3);
      if (ksel < TOPK_) {
        int n = sidx[ksel];
        unsafeAtomicAdd(&agg[base + (size_t)n * 32 + d], c[r]);
      }
    }
  }
}

// x1 = agg/cnt, merge heads -> Xm bf16 [B,T,N,256]
__global__ void merge_kernel(const float* __restrict__ agg,
                             const float* __restrict__ cnt,
                             bf16_t* __restrict__ Xm) {
  size_t i = (size_t)blockIdx.x * 256 + threadIdx.x;
  if (i >= (size_t)KB_ * T_ * N_ * 32) return;
  int d    = (int)(i & 31);
  size_t e = i >> 5;                 // (kbH*T_ + t)*N_ + n
  int n    = (int)(e % N_);
  size_t f = e / N_;
  int t    = (int)(f % T_);
  int kbH  = (int)(f / T_);
  int head = kbH >> 2;               // /B_
  int bb   = kbH & 3;
  float v = agg[i] / cnt[e];
  Xm[((size_t)((bb * T_ + t) * N_ + n)) * 256 + head * 32 + d] = (bf16_t)v;
}

// ---------------------------------------------------------------------------
// Host launcher
// ---------------------------------------------------------------------------
extern "C" void kernel_launch(void* const* d_in, const int* in_sizes, int n_in,
                              void* d_out, int out_size, void* d_ws, size_t ws_size,
                              hipStream_t stream) {
  const float* X   = (const float*)d_in[0];
  const float* STE = (const float*)d_in[1];
  const float* W7  = (const float*)d_in[2];
  const float* b7  = (const float*)d_in[3];
  const float* W8  = (const float*)d_in[4];
  const float* b8  = (const float*)d_in[5];
  const float* W9  = (const float*)d_in[6];
  const float* b9  = (const float*)d_in[7];
  const float* W10 = (const float*)d_in[8];
  const float* b10 = (const float*)d_in[9];
  const float* W11 = (const float*)d_in[10];
  const float* b11 = (const float*)d_in[11];
  const float* node_emb = (const float*)d_in[12];
  float* out = (float*)d_out;

  auto align256 = [](size_t x) { return (x + 255) & ~(size_t)255; };
  char* ws = (char*)d_ws;
  size_t off = 0;
  auto carve = [&](size_t bytes) { char* p = ws + off; off += align256(bytes); return p; };

  bf16_t* Xc     = (bf16_t*)carve((size_t)MROWS_ * 512 * 2);
  bf16_t* WpQKV  = (bf16_t*)carve((size_t)512 * 768 * 2);
  bf16_t* W10p   = (bf16_t*)carve((size_t)256 * 256 * 2);
  bf16_t* W11p   = (bf16_t*)carve((size_t)256 * 256 * 2);
  bf16_t* qbuf   = (bf16_t*)carve((size_t)KB_ * T_ * N_ * 32 * 2);
  bf16_t* kbuf   = (bf16_t*)carve((size_t)KB_ * T_ * N_ * 32 * 2);
  bf16_t* vbuf   = (bf16_t*)carve((size_t)KB_ * T_ * N_ * 32 * 2);
  int*    idxbuf = (int*)   carve((size_t)KB_ * T_ * MEM_ * TOPK_ * 4);
  float*  agg    = (float*) carve((size_t)KB_ * T_ * N_ * 32 * 4);
  float*  cnt    = (float*) carve((size_t)KB_ * T_ * N_ * 4);
  bf16_t* Xm     = (bf16_t*)carve((size_t)MROWS_ * 256 * 2);
  bf16_t* hidden = (bf16_t*)carve((size_t)MROWS_ * 256 * 2);

  // 1. weight packing + input conversion
  pack_qkv_weights<<<(512 * 768 + 255) / 256, 256, 0, stream>>>(W7, W8, W9, WpQKV);
  pack_weight256<<<(256 * 256 + 255) / 256, 256, 0, stream>>>(W10, W10p);
  pack_weight256<<<(256 * 256 + 255) / 256, 256, 0, stream>>>(W11, W11p);
  {
    size_t tot = (size_t)MROWS_ * 512;
    build_xc<<<(unsigned)((tot + 255) / 256), 256, 0, stream>>>(X, STE, Xc);
  }

  // 2. fused QKV GEMM (bias + ReLU, split-head output)
  qkv_gemm_kernel<<<dim3(MROWS_ / 32, 3), 256, 0, stream>>>(
      Xc, WpQKV, b7, b8, b9, qbuf, kbuf, vbuf);

  // 3. scores + top-50 routing
  scores_topk_kernel<<<KB_ * T_ * MEM_, 256, 0, stream>>>(qbuf, kbuf, node_emb, idxbuf);

  // 4. init accumulators
  {
    size_t tot = (size_t)KB_ * T_ * N_ * 32;
    init_accum<<<(unsigned)((tot + 255) / 256), 256, 0, stream>>>(agg, cnt);
  }

  // 5. subgraph attention + scatter-add
  attn_scatter_kernel<<<KB_ * T_ * MEM_, 128, 0, stream>>>(
      qbuf, kbuf, vbuf, idxbuf, agg, cnt);

  // 6. normalize + merge heads
  {
    size_t tot = (size_t)KB_ * T_ * N_ * 32;
    merge_kernel<<<(unsigned)((tot + 255) / 256), 256, 0, stream>>>(agg, cnt, Xm);
  }

  // 7. output MLP: relu(Xm@W10+b10) @ W11 + b11
  gemm_bias_kernel<<<MROWS_ / 32, 256, 0, stream>>>(Xm, W10p, b10, hidden, nullptr, 1);
  gemm_bias_kernel<<<MROWS_ / 32, 256, 0, stream>>>(hidden, W11p, b11, nullptr, out, 0);
}
